// TFAttention_21260088115806
// MI455X (gfx1250) — matrix operand order repaired and checked by reading, not compiled, not run on hardware
//
#include <hip/hip_runtime.h>
#include <hip/hip_bf16.h>

typedef __attribute__((ext_vector_type(4)))  float    v4f;
typedef __attribute__((ext_vector_type(8)))  float    v8f;
typedef __attribute__((ext_vector_type(4)))  _Float16 v4h;
typedef __attribute__((ext_vector_type(8)))  _Float16 v8h;
typedef __attribute__((ext_vector_type(16))) _Float16 v16h;

static __device__ __forceinline__ v16h cat8(v8h lo, v8h hi) {
  v16h r;
#pragma unroll
  for (int i = 0; i < 8; ++i) { r[i] = lo[i]; r[i + 8] = hi[i]; }
  return r;
}

static __device__ __forceinline__ v8f wmma_f16(v16h a, v16h b, v8f c) {
  // D = A(16x32 f16) x B(32x16 f16) + C(16x16 f32)
  return __builtin_amdgcn_wmma_f32_16x16x32_f16(
      /*neg_a=*/false, a, /*neg_b=*/false, b,
      /*c_mod=*/(short)0, c, /*reuse_a=*/false, /*reuse_b=*/false);
}

// ---------------------------------------------------------------------------
// f32 -> f16 elementwise convert (4 elems/thread)
// ---------------------------------------------------------------------------
__global__ __launch_bounds__(256) void cvt_f32_to_f16_kernel(
    const float* __restrict__ in, _Float16* __restrict__ out, int n4) {
  int i = blockIdx.x * blockDim.x + threadIdx.x;
  if (i < n4) {
    v4f v = ((const v4f*)in)[i];
    v4h h;
#pragma unroll
    for (int e = 0; e < 4; ++e) h[e] = (_Float16)v[e];
    ((v4h*)out)[i] = h;
  }
}

// ---------------------------------------------------------------------------
// W[K][N] f32  ->  Wt[N][K] f16  (32x32 LDS tile transpose, block 32x8)
// ---------------------------------------------------------------------------
__global__ __launch_bounds__(256) void transpose_f32_to_f16_kernel(
    const float* __restrict__ W, _Float16* __restrict__ Wt, int K, int N) {
  __shared__ float tile[32][33];
  const int k0 = blockIdx.x * 32;
  const int n0 = blockIdx.y * 32;
#pragma unroll
  for (int i = threadIdx.y; i < 32; i += 8)
    tile[i][threadIdx.x] = W[(long)(k0 + i) * N + n0 + threadIdx.x];
  __syncthreads();
#pragma unroll
  for (int i = threadIdx.y; i < 32; i += 8)
    Wt[(long)(n0 + i) * K + k0 + threadIdx.x] = (_Float16)tile[threadIdx.x][i];
}

// ---------------------------------------------------------------------------
// C[M][N] = A[M][K] * Bt[N][K]^T + bias[N]
// Block tile 64x128, 128 threads (4 waves); wave owns 32x64 (2x4 WMMA accs).
// Double-buffered LDS: fill tile t+1 while computing tile t.
// ---------------------------------------------------------------------------
template <bool HALF_OUT>
__global__ __launch_bounds__(128) void gemm_wmma_kernel(
    const _Float16* __restrict__ A, const _Float16* __restrict__ Bt,
    const float* __restrict__ bias, void* __restrict__ Cout,
    int M, int N, int K) {
  __shared__ _Float16 As[2][64][72];    // 18.4 KB
  __shared__ _Float16 Bs[2][128][72];   // 36.9 KB

  const int tid  = threadIdx.x;
  const int lane = tid & 31;
  const int w    = tid >> 5;
  const int wm   = (w >> 1) * 32;   // 0 / 32
  const int wn   = (w & 1) * 64;    // 0 / 64
  const int half = lane >> 4;
  const int l16  = lane & 15;
  const long mbase = (long)blockIdx.x * 64;
  const long nbase = (long)blockIdx.y * 128;
  (void)M;

  v8f acc[2][4] = {};

  auto load_tiles = [&](int k0, int buf) {
#pragma unroll
    for (int it = 0; it < 4; ++it) {          // A: 64x64 = 512 chunks of 8 f16
      const int c    = tid + it * 128;
      const int row  = c >> 3;
      const int col8 = (c & 7) * 8;
      *(v8h*)&As[buf][row][col8] =
          *(const v8h*)&A[(mbase + row) * (long)K + k0 + col8];
    }
#pragma unroll
    for (int it = 0; it < 8; ++it) {          // B: 128x64 = 1024 chunks
      const int c    = tid + it * 128;
      const int row  = c >> 3;
      const int col8 = (c & 7) * 8;
      *(v8h*)&Bs[buf][row][col8] =
          *(const v8h*)&Bt[(nbase + row) * (long)K + k0 + col8];
    }
  };

  const int T = K >> 6;   // 64-wide K steps
  load_tiles(0, 0);

  for (int t = 0; t < T; ++t) {
    __syncthreads();                       // buf t ready; buf t+1 free
    if (t + 1 < T) load_tiles((t + 1) << 6, (t + 1) & 1);
    if (t + 2 < T) {                       // pull tile t+2 toward GL2/WGP$
      const int row  = tid >> 3;
      const int col8 = (tid & 7) * 8;
      __builtin_prefetch(&A[(mbase + row) * (long)K + ((t + 2) << 6) + col8], 0, 3);
      __builtin_prefetch(&Bt[(nbase + row) * (long)K + ((t + 2) << 6) + col8], 0, 3);
      __builtin_prefetch(&Bt[(nbase + 64 + row) * (long)K + ((t + 2) << 6) + col8], 0, 3);
    }
    const int buf = t & 1;
#pragma unroll
    for (int kk = 0; kk < 64; kk += 32) {
      v16h afr[2], bfr[4];
#pragma unroll
      for (int i = 0; i < 2; ++i) {
        const int r = wm + 16 * i + l16;
        afr[i] = cat8(*(const v8h*)&As[buf][r][kk + 16 * half],
                      *(const v8h*)&As[buf][r][kk + 16 * half + 8]);
      }
#pragma unroll
      for (int j = 0; j < 4; ++j) {
        const int r = wn + 16 * j + l16;
        bfr[j] = cat8(*(const v8h*)&Bs[buf][r][kk + 16 * half],
                      *(const v8h*)&Bs[buf][r][kk + 16 * half + 8]);
      }
#pragma unroll
      for (int i = 0; i < 2; ++i)
#pragma unroll
        for (int j = 0; j < 4; ++j)
          acc[i][j] = wmma_f16(afr[i], bfr[j], acc[i][j]);
    }
  }

#pragma unroll
  for (int j = 0; j < 4; ++j) {
    const long n   = nbase + wn + 16 * j + l16;
    const float bv = bias[n];
#pragma unroll
    for (int i = 0; i < 2; ++i) {
      const long m0 = mbase + wm + 16 * i + 8 * half;
#pragma unroll
      for (int r = 0; r < 8; ++r) {
        const float v  = acc[i][j][r] + bv;
        const long idx = (m0 + r) * (long)N + n;
        if (HALF_OUT) ((_Float16*)Cout)[idx] = (_Float16)v;
        else          ((float*)Cout)[idx]    = v;
      }
    }
  }
}

// ---------------------------------------------------------------------------
// Causal flash attention. qkv: [B*S][3072] f16 (q|k|v, 16 heads x 64).
// grid = (S/64, B*H); block = 128 (4 waves); wave w owns query rows w*16..+15.
// Double-buffered K / V^T tiles; single barrier per key-tile.
// ---------------------------------------------------------------------------
__global__ __launch_bounds__(128) void flash_attn_kernel(
    const _Float16* __restrict__ qkv, _Float16* __restrict__ aout) {
  __shared__ _Float16 Qs[64][72];
  __shared__ _Float16 Ks[2][64][72];
  __shared__ _Float16 Vts[2][64][72];   // transposed: [d][key]
  __shared__ _Float16 Ps[4][16][72];    // per-wave probability tile

  const int tid  = threadIdx.x;
  const int lane = tid & 31;
  const int w    = tid >> 5;
  const int half = lane >> 4;
  const int l16  = lane & 15;
  const int qt = blockIdx.x;
  const int bh = blockIdx.y;
  const int b  = bh >> 4;
  const int h  = bh & 15;

  const long rowQ0 = (long)b * 2048 + (long)qt * 64;
  const _Float16* qb = qkv + rowQ0 * 3072 + h * 64;

#pragma unroll
  for (int it = 0; it < 4; ++it) {
    const int c    = tid + it * 128;
    const int row  = c >> 3;
    const int col8 = (c & 7) * 8;
    *(v8h*)&Qs[row][col8] = *(const v8h*)&qb[(long)row * 3072 + col8];
  }

  auto load_kv = [&](int kt, int buf) {
    const long rowK0 = (long)b * 2048 + (long)kt * 64;
    const _Float16* kb = qkv + rowK0 * 3072 + 1024 + h * 64;
    const _Float16* vb = kb + 1024;
#pragma unroll
    for (int it = 0; it < 4; ++it) {
      const int c    = tid + it * 128;
      const int row  = c >> 3;
      const int col8 = (c & 7) * 8;
      *(v8h*)&Ks[buf][row][col8] = *(const v8h*)&kb[(long)row * 3072 + col8];
      const v8h vv = *(const v8h*)&vb[(long)row * 3072 + col8];
#pragma unroll
      for (int e = 0; e < 8; ++e) Vts[buf][col8 + e][row] = vv[e];
    }
  };

  float m_i[8], l_i[8];
  v8f o[4] = {};
#pragma unroll
  for (int r = 0; r < 8; ++r) { m_i[r] = -3.0e38f; l_i[r] = 0.0f; }

  load_kv(0, 0);

  for (int kt = 0; kt <= qt; ++kt) {
    __syncthreads();                          // buf kt ready; buf kt+1 free
    if (kt + 1 <= qt) load_kv(kt + 1, (kt + 1) & 1);
    const int buf = kt & 1;

    // ---- scores: S[16 x 64] = Q_rows x K_tile^T ----
    v8f sacc[4] = {};
#pragma unroll
    for (int kk = 0; kk < 64; kk += 32) {
      const int qr = w * 16 + l16;
      const v16h aq = cat8(*(const v8h*)&Qs[qr][kk + 16 * half],
                           *(const v8h*)&Qs[qr][kk + 16 * half + 8]);
#pragma unroll
      for (int j = 0; j < 4; ++j) {
        const int kr = j * 16 + l16;
        const v16h bk = cat8(*(const v8h*)&Ks[buf][kr][kk + 16 * half],
                             *(const v8h*)&Ks[buf][kr][kk + 16 * half + 8]);
        sacc[j] = wmma_f16(aq, bk, sacc[j]);
      }
    }

    // ---- online softmax (per query row; rows private to this wave) ----
    const bool diag = (kt == qt);
    float p[4][8];
#pragma unroll
    for (int r = 0; r < 8; ++r) {
      const int qrow = qt * 64 + w * 16 + r + 8 * half;
      float mx = -3.0e38f;
#pragma unroll
      for (int j = 0; j < 4; ++j) {
        float s = sacc[j][r] * 0.125f;               // 1/sqrt(64)
        const int krow = kt * 64 + j * 16 + l16;
        if (diag && krow > qrow) s = -10000.0f;      // MASK_VAL, matches ref
        p[j][r] = s;
        mx = fmaxf(mx, s);
      }
      mx = fmaxf(mx, __shfl_xor(mx, 1, 32));
      mx = fmaxf(mx, __shfl_xor(mx, 2, 32));
      mx = fmaxf(mx, __shfl_xor(mx, 4, 32));
      mx = fmaxf(mx, __shfl_xor(mx, 8, 32));
      const float mnew  = fmaxf(m_i[r], mx);
      const float alpha = __expf(m_i[r] - mnew);
      float rs = 0.0f;
#pragma unroll
      for (int j = 0; j < 4; ++j) {
        const float e = __expf(p[j][r] - mnew);
        p[j][r] = e;
        rs += e;
      }
      rs += __shfl_xor(rs, 1, 32);
      rs += __shfl_xor(rs, 2, 32);
      rs += __shfl_xor(rs, 4, 32);
      rs += __shfl_xor(rs, 8, 32);
      l_i[r] = l_i[r] * alpha + rs;
      m_i[r] = mnew;
#pragma unroll
      for (int j = 0; j < 4; ++j) o[j][r] *= alpha;
    }

    // ---- P: C-fragment layout -> A-fragment layout via per-wave LDS tile.
    // Same-wave DS ops are in-order and gated by s_wait_dscnt; no barrier.
#pragma unroll
    for (int j = 0; j < 4; ++j)
#pragma unroll
      for (int r = 0; r < 8; ++r)
        Ps[w][r + 8 * half][j * 16 + l16] = (_Float16)p[j][r];

    // ---- O += P x V ----
#pragma unroll
    for (int kk = 0; kk < 64; kk += 32) {
      const v16h ap = cat8(*(const v8h*)&Ps[w][l16][kk + 16 * half],
                           *(const v8h*)&Ps[w][l16][kk + 16 * half + 8]);
#pragma unroll
      for (int j = 0; j < 4; ++j) {
        const int vr = j * 16 + l16;
        const v16h bv = cat8(*(const v8h*)&Vts[buf][vr][kk + 16 * half],
                             *(const v8h*)&Vts[buf][vr][kk + 16 * half + 8]);
        o[j] = wmma_f16(ap, bv, o[j]);
      }
    }
  }

  // ---- finalize: O /= l, merged-head store ----
#pragma unroll
  for (int r = 0; r < 8; ++r) {
    const float inv  = 1.0f / l_i[r];
    const long  srow = rowQ0 + w * 16 + r + 8 * half;
#pragma unroll
    for (int j = 0; j < 4; ++j) {
      const int col = h * 64 + j * 16 + l16;
      aout[srow * 1024 + col] = (_Float16)(o[j][r] * inv);
    }
  }
}

// ---------------------------------------------------------------------------
extern "C" void kernel_launch(void* const* d_in, const int* in_sizes, int n_in,
                              void* d_out, int out_size, void* d_ws, size_t ws_size,
                              hipStream_t stream) {
  (void)in_sizes; (void)n_in; (void)out_size; (void)ws_size;
  const float* x        = (const float*)d_in[0];  // [2,2048,1024]
  const float* c_attn_w = (const float*)d_in[1];  // [1024,3072]
  const float* c_attn_b = (const float*)d_in[2];  // [3072]
  const float* c_proj_w = (const float*)d_in[3];  // [1024,1024]
  const float* c_proj_b = (const float*)d_in[4];  // [1024]
  float* out = (float*)d_out;                     // [2,2048,1024]

  const long M = 4096, D = 1024, N3 = 3072;

  size_t off = 0;
  auto take = [&](size_t bytes) {
    void* p = (char*)d_ws + off;
    off += (bytes + 255) & ~(size_t)255;
    return p;
  };
  _Float16* Xh     = (_Float16*)take((size_t)M * D * 2);    //  8 MB
  _Float16* WqkvT  = (_Float16*)take((size_t)N3 * D * 2);   //  6 MB
  _Float16* WprojT = (_Float16*)take((size_t)D * D * 2);    //  2 MB
  _Float16* QKVh   = (_Float16*)take((size_t)M * N3 * 2);   // 24 MB
  _Float16* Ah     = (_Float16*)take((size_t)M * D * 2);    //  8 MB

  // 1) convert x -> f16
  {
    const int n4 = (int)(M * D / 4);
    cvt_f32_to_f16_kernel<<<(n4 + 255) / 256, 256, 0, stream>>>(x, Xh, n4);
  }
  // 2) transpose+convert weights
  transpose_f32_to_f16_kernel<<<dim3(1024 / 32, 3072 / 32), dim3(32, 8), 0, stream>>>(
      c_attn_w, WqkvT, 1024, 3072);
  transpose_f32_to_f16_kernel<<<dim3(1024 / 32, 1024 / 32), dim3(32, 8), 0, stream>>>(
      c_proj_w, WprojT, 1024, 1024);
  // 3) QKV = X @ Wqkv + b   (f16 out)
  gemm_wmma_kernel<true><<<dim3(64, 24), 128, 0, stream>>>(
      Xh, WqkvT, c_attn_b, (void*)QKVh, 4096, 3072, 1024);
  // 4) causal flash attention -> merged-head A (f16)
  flash_attn_kernel<<<dim3(32, 32), 128, 0, stream>>>(QKVh, Ah);
  // 5) out = A @ Wproj + b  (f32 out)
  gemm_wmma_kernel<false><<<dim3(64, 8), 128, 0, stream>>>(
      Ah, WprojT, c_proj_b, (void*)out, 4096, 1024, 1024);
}
